// DecoderRNN_54838142436015
// MI455X (gfx1250) — matrix operand order repaired and checked by reading, not compiled, not run on hardware
//
#include <hip/hip_runtime.h>
#include <stdint.h>

// Problem dims (match reference)
#define T_STEPS 8192
#define IN_DIM  2048
#define H_DIM   1024
#define G4      4096      // 4*H
#define MID_DIM 128
#define RWG     64        // workgroups in the persistent recurrence kernel

typedef uint16_t u16;
typedef __attribute__((ext_vector_type(16))) __bf16 v16bf;
typedef __attribute__((ext_vector_type(8)))  float  v8f;

struct alignas(16) U4 { unsigned x, y, z, w; };
struct alignas(16) F4 { float x, y, z, w; };
union FragU { U4 q[2]; v16bf v; };

__device__ __forceinline__ float bflo(unsigned u) { return __uint_as_float(u << 16); }
__device__ __forceinline__ float bfhi(unsigned u) { return __uint_as_float(u & 0xffff0000u); }

__device__ __forceinline__ u16 f2bf(float f) {          // round-to-nearest-even
  unsigned u = __float_as_uint(f);
  unsigned r = 0x7fffu + ((u >> 16) & 1u);
  return (u16)((u + r) >> 16);
}

// ---------------------------------------------------------------- fp32 -> bf16
__global__ void f2bf_kernel(const float* __restrict__ in, u16* __restrict__ out, int n) {
  int i = blockIdx.x * blockDim.x + threadIdx.x;
  int s = gridDim.x * blockDim.x;
  for (; i < n; i += s) out[i] = f2bf(in[i]);
}

// ------------------------------------------- WMMA GEMM: xW = x @ W_ih^T + bias
// A = x_seq bf16 [8192,2048] row-major, B = W_ih bf16 [4096,2048] row-major.
// C[m][n] = sum_k A[m][k]*B[n][k] + (b_ih[n]+b_hh[n]).
// WG = 256 thr = 8 waves (2x4); wave tile 32x32 (2x2 wmma frags); WG tile 64x128.
__global__ __launch_bounds__(256) void gemm_xw_kernel(
    const u16* __restrict__ A, const u16* __restrict__ B,
    const float* __restrict__ b_ih, const float* __restrict__ b_hh,
    float* __restrict__ C) {
  const int K = IN_DIM, N = G4;
  const int lane = threadIdx.x & 31;
  const int wave = threadIdx.x >> 5;
  const int wm = wave >> 2, wn = wave & 3;
  const int m0 = blockIdx.y * 64 + wm * 32;
  const int n0 = blockIdx.x * 128 + wn * 32;
  const int lr = lane & 15;
  const int lh = lane >> 4;

  // A frag (16x32 bf16, contiguous K order, same as B): lane holds row (lane&15); 16 contiguous K at lh*16
  const u16* a0p = A + (size_t)(m0 + lr) * K + lh * 16;
  const u16* a1p = a0p + (size_t)16 * K;
  // B frag (32x16 bf16, contiguous K order, same as A): lane holds col (lane&15); 16 contiguous K at lh*16
  const u16* b0p = B + (size_t)(n0 + lr) * K + lh * 16;
  const u16* b1p = b0p + (size_t)16 * K;

  v8f c00 = {}, c01 = {}, c10 = {}, c11 = {};
  for (int k = 0; k < K; k += 32) {
    FragU a0, a1, b0, b1;
    a0.q[0] = *(const U4*)(a0p + k); a0.q[1] = *(const U4*)(a0p + k + 8);
    a1.q[0] = *(const U4*)(a1p + k); a1.q[1] = *(const U4*)(a1p + k + 8);
    b0.q[0] = *(const U4*)(b0p + k); b0.q[1] = *(const U4*)(b0p + k + 8);
    b1.q[0] = *(const U4*)(b1p + k); b1.q[1] = *(const U4*)(b1p + k + 8);
    c00 = __builtin_amdgcn_wmma_f32_16x16x32_bf16(false, a0.v, false, b0.v, (short)0, c00, false, false);
    c01 = __builtin_amdgcn_wmma_f32_16x16x32_bf16(false, a0.v, false, b1.v, (short)0, c01, false, false);
    c10 = __builtin_amdgcn_wmma_f32_16x16x32_bf16(false, a1.v, false, b0.v, (short)0, c10, false, false);
    c11 = __builtin_amdgcn_wmma_f32_16x16x32_bf16(false, a1.v, false, b1.v, (short)0, c11, false, false);
  }

  v8f acc[2][2] = { { c00, c01 }, { c10, c11 } };
  #pragma unroll
  for (int fm = 0; fm < 2; ++fm) {
    #pragma unroll
    for (int fn = 0; fn < 2; ++fn) {
      const int col = n0 + fn * 16 + lr;          // lane's column (C layout)
      const float bias = b_ih[col] + b_hh[col];
      const int rbase = m0 + fm * 16 + lh * 8;    // VGPR r -> row rbase+r
      v8f cc = acc[fm][fn];
      #pragma unroll
      for (int r = 0; r < 8; ++r)
        C[(size_t)(rbase + r) * N + col] = cc[r] + bias;
    }
  }
}

// -------------------------------------------------- persistent LSTM recurrence
// 64 WGs x 256 thr. Each WG owns 16 hidden units (64 gate rows of W_hh, bf16,
// 128 KB) staged in LDS once. Per step: h (4KB) -> LDS, 256-wide dots from LDS,
// shuffle-reduce, gate math, write h to ping-pong buffer, grid barrier.
__global__ __launch_bounds__(256) void lstm_seq_kernel(
    const u16* __restrict__ Whh,     // [4096,1024] bf16
    const float* __restrict__ xW,    // [8192,4096] fp32 (bias folded in)
    float* __restrict__ hbuf,        // [2][1024] ping-pong
    unsigned* __restrict__ counter) {
  __shared__ u16  wlds[64 * H_DIM];  // 128 KB (fits CDNA5 320KB/WGP, 2 WGs/WGP)
  __shared__ float hlds[H_DIM];
  const int tid = threadIdx.x;
  const int wg  = blockIdx.x;

  // Stage this WG's 64 W_hh rows into LDS (row = u*4+g  <-  global row g*H + wg*16+u)
  for (int i = tid; i < 64 * (H_DIM / 8); i += 256) {
    int row = i >> 7;                 // H_DIM/8 == 128 chunks per row
    int k8  = (i & 127) << 3;
    int u = row >> 2, g = row & 3;
    int grow = g * H_DIM + wg * 16 + u;
    *(U4*)&wlds[row * H_DIM + k8] = *(const U4*)&Whh[(size_t)grow * H_DIM + k8];
  }
  __syncthreads();

  const int u     = tid >> 4;        // hidden unit within WG (0..15)
  const int sub   = tid & 15;        // 16 threads per unit
  const int g     = sub >> 2;        // gate 0..3 (i,f,g,o)
  const int chunk = sub & 3;         // K chunk 0..3 (256 each)
  const int j     = wg * 16 + u;     // global hidden unit
  const int lane  = tid & 31;
  const int lbase = lane & 16;       // unit's base lane within wave32
  const u16* wrow = &wlds[(u * 4 + g) * H_DIM + chunk * 256];
  float cstate = 0.f;

  for (int t = 0; t < T_STEPS; ++t) {
    const float* hin  = hbuf + (t & 1) * H_DIM;
    float*       hout = hbuf + ((t + 1) & 1) * H_DIM;

    ((F4*)hlds)[tid] = ((const F4*)hin)[tid];   // 256*16B = 4KB broadcast
    __syncthreads();

    float acc = 0.f;
    const float* hp = &hlds[chunk * 256];
    #pragma unroll 4
    for (int k = 0; k < 256; k += 8) {
      U4 wq = *(const U4*)(wrow + k);           // 8 bf16 from LDS
      F4 h0 = *(const F4*)(hp + k);
      F4 h1 = *(const F4*)(hp + k + 4);
      acc = fmaf(bflo(wq.x), h0.x, acc);
      acc = fmaf(bfhi(wq.x), h0.y, acc);
      acc = fmaf(bflo(wq.y), h0.z, acc);
      acc = fmaf(bfhi(wq.y), h0.w, acc);
      acc = fmaf(bflo(wq.z), h1.x, acc);
      acc = fmaf(bfhi(wq.z), h1.y, acc);
      acc = fmaf(bflo(wq.w), h1.z, acc);
      acc = fmaf(bfhi(wq.w), h1.w, acc);
    }
    // reduce across the 4 K-chunks (lane bits 0..1)
    acc += __shfl_xor(acc, 1, 32);
    acc += __shfl_xor(acc, 2, 32);

    float pre = acc + xW[(size_t)t * G4 + g * H_DIM + j];
    float val = (g == 2) ? tanhf(pre) : (1.f / (1.f + __expf(-pre)));
    // gather i,f,g,o from the unit's lanes base+0/4/8/12
    float iv = __shfl(val, lbase + 0,  32);
    float fv = __shfl(val, lbase + 4,  32);
    float gv = __shfl(val, lbase + 8,  32);
    float ov = __shfl(val, lbase + 12, 32);
    if (sub == 0) {
      cstate = fv * cstate + iv * gv;           // c kept in a register forever
      hout[j] = ov * tanhf(cstate);
    }
    __syncthreads();
    __threadfence();
    if (tid == 0) {                             // grid-wide release/acquire barrier
      __hip_atomic_fetch_add(counter, 1u, __ATOMIC_RELEASE, __HIP_MEMORY_SCOPE_AGENT);
      const unsigned target = (unsigned)RWG * (unsigned)(t + 1);
      while (__hip_atomic_load(counter, __ATOMIC_ACQUIRE, __HIP_MEMORY_SCOPE_AGENT) < target)
        __builtin_amdgcn_s_sleep(2);
    }
    __syncthreads();
  }
}

// ---------------------------------------------------------------- classifier
__global__ __launch_bounds__(128) void head_kernel(
    const float* __restrict__ h, const float* __restrict__ W1,
    const float* __restrict__ b1, const float* __restrict__ W2,
    const float* __restrict__ b2, float* __restrict__ out) {
  __shared__ float red[MID_DIM];
  const int m = threadIdx.x;
  float acc = 0.f;
  const float* wr = W1 + (size_t)m * H_DIM;
  for (int k = 0; k < H_DIM; k += 4) {
    F4 hv = *(const F4*)(h + k);
    F4 wv = *(const F4*)(wr + k);
    acc += hv.x * wv.x + hv.y * wv.y + hv.z * wv.z + hv.w * wv.w;
  }
  acc += b1[m];
  acc = fmaxf(acc, 0.f);
  red[m] = acc * W2[m];
  __syncthreads();
  for (int s = 64; s > 0; s >>= 1) {
    if (m < s) red[m] += red[m + s];
    __syncthreads();
  }
  if (m == 0) out[0] = 1.f / (1.f + __expf(-(red[0] + b2[0])));
}

// ------------------------------------------------------------------ launcher
extern "C" void kernel_launch(void* const* d_in, const int* in_sizes, int n_in,
                              void* d_out, int out_size, void* d_ws, size_t ws_size,
                              hipStream_t stream) {
  const float* x    = (const float*)d_in[0];
  const float* Wih  = (const float*)d_in[1];
  const float* Whh  = (const float*)d_in[2];
  const float* b_ih = (const float*)d_in[3];
  const float* b_hh = (const float*)d_in[4];
  const float* W1   = (const float*)d_in[5];
  const float* b1   = (const float*)d_in[6];
  const float* W2   = (const float*)d_in[7];
  const float* b2   = (const float*)d_in[8];
  float* out = (float*)d_out;

  // Workspace layout (~193 MB total)
  char* ws = (char*)d_ws;
  float*    xW    = (float*)(ws);                       // 8192*4096*4 = 128 MB
  u16*      xbf   = (u16*)  (ws + 134217728UL);         // 32 MB
  u16*      wihbf = (u16*)  (ws + 167772160UL);         // 16 MB
  u16*      whhbf = (u16*)  (ws + 184549376UL);         // 8 MB
  float*    hbuf  = (float*)(ws + 192937984UL);         // 2*1024*4 = 8 KB
  unsigned* cnt   = (unsigned*)(ws + 192937984UL + 8192UL);

  // Reset h ping-pong + barrier counter every call (deterministic, capturable)
  hipMemsetAsync(hbuf, 0, 8192 + 256, stream);

  f2bf_kernel<<<2048, 256, 0, stream>>>(x,   xbf,   T_STEPS * IN_DIM);
  f2bf_kernel<<<2048, 256, 0, stream>>>(Wih, wihbf, G4 * IN_DIM);
  f2bf_kernel<<<2048, 256, 0, stream>>>(Whh, whhbf, G4 * H_DIM);

  gemm_xw_kernel<<<dim3(G4 / 128, T_STEPS / 64), 256, 0, stream>>>(
      xbf, wihbf, b_ih, b_hh, xW);

  lstm_seq_kernel<<<RWG, 256, 0, stream>>>(whhbf, xW, hbuf, cnt);

  head_kernel<<<1, MID_DIM, 0, stream>>>(hbuf + (T_STEPS & 1) * H_DIM,
                                         W1, b1, W2, b2, out);
}
